// NeighborhoodSelfAttentionBlock_83219286327330
// MI455X (gfx1250) — matrix operand order repaired and checked by reading, not compiled, not run on hardware
//
#include <hip/hip_runtime.h>
#include <hip/hip_bf16.h>

// ---------------- types for WMMA ----------------
typedef __attribute__((ext_vector_type(16))) __bf16 v16bf;
typedef __attribute__((ext_vector_type(8)))  __bf16 v8bf;
typedef __attribute__((ext_vector_type(8)))  float  v8f;

// ---------------- problem constants ----------------
constexpr int Bn  = 2;
constexpr int Cn  = 512;
constexpr int Hn  = 48;
constexpr int Wn  = 48;
constexpr int HWn = Hn * Wn;          // 2304
constexpr int Pn  = Bn * HWn;         // 4608 pixels
constexpr int Kc  = Cn * 9;           // 4608 (im2col K)
constexpr int N1  = 3 * Cn;           // 1536
constexpr int N2  = Cn;               // 512
constexpr int NH  = 8;
constexpr int DH  = 64;
constexpr int KS  = 7;

// ---------------- helpers ----------------
__device__ __forceinline__ __bf16 f2bf(float f) {
    union { float f; unsigned u; } uf; uf.f = f;
    unsigned u = uf.u;
    unsigned r = (u + 0x7fffu + ((u >> 16) & 1u)) >> 16;   // RNE
    union { unsigned short s; __bf16 b; } ub; ub.s = (unsigned short)r;
    return ub.b;
}

// CDNA5 async global->LDS copy (16 bytes per lane), tracked by ASYNCcnt.
// Generic pointers to __shared__ carry the LDS byte offset in their low 32 bits.
__device__ __forceinline__ void async_copy_b128(const __bf16* gsrc, const __bf16* ldst) {
    unsigned loff = (unsigned)(unsigned long long)ldst;
    asm volatile("global_load_async_to_lds_b128 %0, %1, off"
                 :: "v"(loff), "v"(gsrc) : "memory");
}
__device__ __forceinline__ void wait_async0() {
    asm volatile("s_wait_asynccnt 0x0" ::: "memory");
}

// ================= GroupNorm -> channels-last bf16 =================
__global__ __launch_bounds__(256)
void gn_kernel(const float* __restrict__ x, const float* __restrict__ gw,
               const float* __restrict__ gb, __bf16* __restrict__ hcl) {
    const int b = blockIdx.x >> 4;
    const int g = blockIdx.x & 15;
    const int CG = 32;                       // channels per group
    const int CNT = CG * HWn;                // 73728
    const float* base = x + (size_t)(b * Cn + g * CG) * HWn;

    float s = 0.f, ss = 0.f;
    for (int i = threadIdx.x; i < CNT; i += 256) {
        float v = base[i];
        s += v; ss += v * v;
    }
    __shared__ float sb[256], qb[256];
    sb[threadIdx.x] = s; qb[threadIdx.x] = ss;
    __syncthreads();
    for (int off = 128; off > 0; off >>= 1) {
        if (threadIdx.x < off) {
            sb[threadIdx.x] += sb[threadIdx.x + off];
            qb[threadIdx.x] += qb[threadIdx.x + off];
        }
        __syncthreads();
    }
    const float mean = sb[0] / (float)CNT;
    const float var  = qb[0] / (float)CNT - mean * mean;
    const float inv  = rsqrtf(var + 1e-5f);

    for (int i = threadIdx.x; i < CNT; i += 256) {
        int ci  = i / HWn;
        int rem = i - ci * HWn;
        int c   = g * CG + ci;
        float v = (base[i] - mean) * inv * gw[c] + gb[c];
        hcl[(size_t)(b * HWn + rem) * Cn + c] = f2bf(v);
    }
}

// ========== weight reorder: [O][I][3][3] fp32 -> [O][(t,c)] bf16 ==========
__global__ __launch_bounds__(256)
void wtrans_kernel(const float* __restrict__ w, __bf16* __restrict__ wb, int O) {
    int idx = blockIdx.x * 256 + threadIdx.x;
    if (idx >= O * Kc) return;
    int o = idx / Kc;
    int k = idx - o * Kc;
    int t = k >> 9;            // tap 0..8
    int c = k & 511;
    wb[idx] = f2bf(w[((size_t)o * Cn + c) * 9 + t]);
}

// ========== im2col: src channels-last bf16 [P][512] -> A [P][4608] ==========
__global__ __launch_bounds__(256)
void im2col_kernel(const __bf16* __restrict__ src, __bf16* __restrict__ A) {
    int idx = blockIdx.x * 256 + threadIdx.x;        // over Pn * (Kc/8)
    if (idx >= Pn * (Kc / 8)) return;
    int p  = idx / (Kc / 8);
    int kk = (idx - p * (Kc / 8)) * 8;
    int t  = kk >> 9;
    int c  = kk & 511;
    int b   = p / HWn;
    int rem = p - b * HWn;
    int y   = rem / Wn;
    int xw  = rem - y * Wn;
    int ny  = y + t / 3 - 1;
    int nx  = xw + t % 3 - 1;
    uint4 val = make_uint4(0u, 0u, 0u, 0u);
    if ((unsigned)ny < (unsigned)Hn && (unsigned)nx < (unsigned)Wn)
        val = *(const uint4*)(src + ((size_t)(b * HWn + ny * Wn + nx) * Cn + c));
    *(uint4*)(A + (size_t)idx * 8) = val;
}

// ================= bf16 WMMA GEMM with async-LDS double buffering =================
// D[P][Ncols] = A[P][Kc] * B[Ncols][Kc]^T
// mode 0: out fp32 channels-last, + bias[col]
// mode 1: out NCHW,               + bias[col] + skip (residual), col==channel
constexpr int LPAD = 40;   // 32 K-elems + pad -> 80B rows: 16B aligned, bank stride 20 dwords
__global__ __launch_bounds__(256)
void gemm_bf16_kernel(const __bf16* __restrict__ A, const __bf16* __restrict__ Bm,
                      const float* __restrict__ bias, const float* __restrict__ skip,
                      float* __restrict__ out, int Ncols, int mode) {
    __shared__ __bf16 lA[2][64][LPAD];
    __shared__ __bf16 lB[2][256][LPAD];

    const int tid   = threadIdx.x;
    const int lane  = tid & 31;
    const int wave  = tid >> 5;
    const int l16   = lane & 15;
    const int lhalf = lane >> 4;                         // 0 | 1
    const int m_wg  = blockIdx.x * 64;
    const int n_wg  = blockIdx.y * 256;

    // async-fill assignment: each thread copies 1 A row-seg + 4 B row-segs (16B each)
    const int lrow = tid >> 2;                           // 0..63
    const int lseg = (tid & 3) * 8;                      // bf16 elements
    const __bf16* gA  = A  + (size_t)(m_wg + lrow) * Kc + lseg;
    const __bf16* gB0 = Bm + (size_t)(n_wg + lrow) * Kc + lseg;

    // fragment coordinates
    const int rowA0 = (wave & 1) * 32 + l16;             // + mi*16
    const int colB0 = (wave >> 1) * 64 + l16;            // + ni*16

    const v8f vzero = {0.f,0.f,0.f,0.f,0.f,0.f,0.f,0.f};
    v8f acc[2][4];
#pragma unroll
    for (int mi = 0; mi < 2; ++mi)
#pragma unroll
        for (int ni = 0; ni < 4; ++ni) acc[mi][ni] = vzero;

    // prologue: async-load tile 0 into buffer 0
    {
        async_copy_b128(gA, &lA[0][lrow][lseg]);
#pragma unroll
        for (int i = 0; i < 4; ++i)
            async_copy_b128(gB0 + (size_t)(64 * i) * Kc, &lB[0][lrow + 64 * i][lseg]);
    }

    const int KT = Kc / 32;                              // 144
    for (int kt = 0; kt < KT; ++kt) {
        const int buf = kt & 1;
        wait_async0();        // my async loads for `buf` complete
        __syncthreads();      // everyone's loads complete -> tile visible

        if (kt + 1 < KT) {    // start filling the other buffer
            const int nb = buf ^ 1;
            const int k0n = (kt + 1) * 32;
            async_copy_b128(gA + k0n, &lA[nb][lrow][lseg]);
#pragma unroll
            for (int i = 0; i < 4; ++i)
                async_copy_b128(gB0 + (size_t)(64 * i) * Kc + k0n,
                                &lB[nb][lrow + 64 * i][lseg]);
        }

        // fragments from LDS (ds_load_b128)
        union { v16bf v; v8bf h[2]; } ua;
        v16bf afr[2], bfr[4];
#pragma unroll
        for (int mi = 0; mi < 2; ++mi) {
            ua.h[0] = *(const v8bf*)&lA[buf][rowA0 + mi * 16][lhalf * 8];
            ua.h[1] = *(const v8bf*)&lA[buf][rowA0 + mi * 16][16 + lhalf * 8];
            afr[mi] = ua.v;
        }
#pragma unroll
        for (int ni = 0; ni < 4; ++ni) {
            ua.h[0] = *(const v8bf*)&lB[buf][colB0 + ni * 16][lhalf * 8];
            ua.h[1] = *(const v8bf*)&lB[buf][colB0 + ni * 16][16 + lhalf * 8];
            bfr[ni] = ua.v;
        }
#pragma unroll
        for (int ni = 0; ni < 4; ++ni) {
            acc[0][ni] = __builtin_amdgcn_wmma_f32_16x16x32_bf16(
                false, afr[0], false, bfr[ni], (short)0, acc[0][ni], false, false);
            acc[1][ni] = __builtin_amdgcn_wmma_f32_16x16x32_bf16(
                false, afr[1], false, bfr[ni], (short)0, acc[1][ni], false, false);
        }
        __syncthreads();      // done reading `buf` before it is refilled
    }

    // epilogue: C/D layout -> lane holds N = l16, VGPR r holds M = r + 8*lhalf
    const int m_base = m_wg + (wave & 1) * 32;
    const int n_base = n_wg + (wave >> 1) * 64;
#pragma unroll
    for (int mi = 0; mi < 2; ++mi) {
#pragma unroll
        for (int ni = 0; ni < 4; ++ni) {
#pragma unroll
            for (int r = 0; r < 8; ++r) {
                int m = m_base + mi * 16 + r + lhalf * 8;
                int n = n_base + ni * 16 + l16;
                float v = acc[mi][ni][r] + bias[n];
                if (mode == 0) {
                    out[(size_t)m * Ncols + n] = v;
                } else {
                    int b   = m / HWn;
                    int rem = m - b * HWn;
                    size_t idx = (size_t)(b * Cn + n) * HWn + rem;
                    out[idx] = v + skip[idx];
                }
            }
        }
    }
}

// ========== cosine scale on q and k (in-place, qkv fp32 [P][1536]) ==========
__global__ __launch_bounds__(256)
void cosine_kernel(float* __restrict__ qkv, const float* __restrict__ scale) {
    const int lane = threadIdx.x & 31;
    const int wid  = blockIdx.x * 8 + (threadIdx.x >> 5);   // over Pn*NH*2
    const int which = wid & 1;                              // 0=q, 1=k
    const int head  = (wid >> 1) & (NH - 1);
    const int p     = wid >> 4;
    float* v = qkv + (size_t)p * N1 + which * Cn + head * DH;
    float2 t = *(float2*)(v + lane * 2);
    float ssq = t.x * t.x + t.y * t.y;
#pragma unroll
    for (int off = 16; off > 0; off >>= 1)
        ssq += __shfl_xor(ssq, off, 32);
    float f = sqrtf(scale[head]) * rsqrtf(ssq + 1e-6f);
    t.x *= f; t.y *= f;
    *(float2*)(v + lane * 2) = t;
}

// ========== 7x7 neighborhood attention, one wave per (pixel, head) ==========
__global__ __launch_bounds__(256)
void natten_kernel(const float* __restrict__ qkv, __bf16* __restrict__ ocl) {
    __shared__ float qbuf[8][64];
    __shared__ float abuf[8][52];
    const int wave = threadIdx.x >> 5;
    const int lane = threadIdx.x & 31;
    const int wid  = blockIdx.x * 8 + wave;                 // over Pn*NH
    const int head = wid & (NH - 1);
    const int p    = wid >> 3;
    const int b    = p / HWn;
    const int rem  = p - b * HWn;
    const int y    = rem / Wn;
    const int xw   = rem - y * Wn;
    const int y0   = min(max(y - KS / 2, 0), Hn - KS);
    const int x0   = min(max(xw - KS / 2, 0), Wn - KS);

    const float* qp = qkv + (size_t)p * N1 + head * DH;
    float2 q2 = *(const float2*)(qp + lane * 2);
    qbuf[wave][lane * 2]     = q2.x;
    qbuf[wave][lane * 2 + 1] = q2.y;
    __syncthreads();

    const float* qs = qbuf[wave];
    const int j0 = lane, j1 = lane + 32;
    float d0 = -1e30f, d1 = -1e30f;
    {
        int ny = y0 + j0 / KS, nx = x0 + j0 % KS;
        const float* kv = qkv + (size_t)(b * HWn + ny * Wn + nx) * N1 + Cn + head * DH;
        float s = 0.f;
#pragma unroll
        for (int d = 0; d < DH; d += 4) {
            float4 kk = *(const float4*)(kv + d);
            s += qs[d] * kk.x + qs[d + 1] * kk.y + qs[d + 2] * kk.z + qs[d + 3] * kk.w;
        }
        d0 = s;
    }
    if (j1 < KS * KS) {
        int ny = y0 + j1 / KS, nx = x0 + j1 % KS;
        const float* kv = qkv + (size_t)(b * HWn + ny * Wn + nx) * N1 + Cn + head * DH;
        float s = 0.f;
#pragma unroll
        for (int d = 0; d < DH; d += 4) {
            float4 kk = *(const float4*)(kv + d);
            s += qs[d] * kk.x + qs[d + 1] * kk.y + qs[d + 2] * kk.z + qs[d + 3] * kk.w;
        }
        d1 = s;
    }
    float mx = fmaxf(d0, d1);
#pragma unroll
    for (int off = 16; off > 0; off >>= 1)
        mx = fmaxf(mx, __shfl_xor(mx, off, 32));
    float e0 = __expf(d0 - mx);
    float e1 = (j1 < KS * KS) ? __expf(d1 - mx) : 0.f;
    float sum = e0 + e1;
#pragma unroll
    for (int off = 16; off > 0; off >>= 1)
        sum += __shfl_xor(sum, off, 32);
    float inv = 1.f / sum;
    abuf[wave][j0] = e0 * inv;
    if (j1 < KS * KS) abuf[wave][j1] = e1 * inv;
    __syncthreads();

    float o0 = 0.f, o1 = 0.f;
#pragma unroll 7
    for (int j = 0; j < KS * KS; ++j) {
        float a = abuf[wave][j];
        int ny = y0 + j / KS, nx = x0 + j % KS;
        const float* vv = qkv + (size_t)(b * HWn + ny * Wn + nx) * N1 + 2 * Cn + head * DH;
        o0 += a * vv[lane];
        o1 += a * vv[lane + 32];
    }
    ocl[(size_t)p * Cn + head * DH + lane]      = f2bf(o0);
    ocl[(size_t)p * Cn + head * DH + lane + 32] = f2bf(o1);
}

// ---------------- workspace layout (bytes) ----------------
static constexpr size_t AL = 256;
static constexpr size_t alup(size_t x) { return (x + AL - 1) & ~(AL - 1); }
static constexpr size_t OFF_HCL = 0;
static constexpr size_t OFF_A   = alup(OFF_HCL + (size_t)Pn * Cn * 2);
static constexpr size_t OFF_WQ  = alup(OFF_A   + (size_t)Pn * Kc * 2);
static constexpr size_t OFF_WO  = alup(OFF_WQ  + (size_t)N1 * Kc * 2);
static constexpr size_t OFF_QKV = alup(OFF_WO  + (size_t)N2 * Kc * 2);
static constexpr size_t OFF_OCL = alup(OFF_QKV + (size_t)Pn * N1 * 4);

extern "C" void kernel_launch(void* const* d_in, const int* in_sizes, int n_in,
                              void* d_out, int out_size, void* d_ws, size_t ws_size,
                              hipStream_t stream) {
    const float* x     = (const float*)d_in[0];
    const float* gn_w  = (const float*)d_in[1];
    const float* gn_b  = (const float*)d_in[2];
    const float* qkv_w = (const float*)d_in[3];
    const float* qkv_b = (const float*)d_in[4];
    const float* scale = (const float*)d_in[5];
    const float* out_w = (const float*)d_in[6];
    const float* out_b = (const float*)d_in[7];
    float* y = (float*)d_out;

    char* ws = (char*)d_ws;
    __bf16* hcl = (__bf16*)(ws + OFF_HCL);
    __bf16* A   = (__bf16*)(ws + OFF_A);
    __bf16* Wq  = (__bf16*)(ws + OFF_WQ);
    __bf16* Wo  = (__bf16*)(ws + OFF_WO);
    float*  qkv = (float*)(ws + OFF_QKV);
    __bf16* ocl = (__bf16*)(ws + OFF_OCL);

    // 1. GroupNorm -> channels-last bf16
    gn_kernel<<<Bn * 16, 256, 0, stream>>>(x, gn_w, gn_b, hcl);

    // 2. weight reorders
    wtrans_kernel<<<(N1 * Kc + 255) / 256, 256, 0, stream>>>(qkv_w, Wq, N1);
    wtrans_kernel<<<(N2 * Kc + 255) / 256, 256, 0, stream>>>(out_w, Wo, N2);

    // 3. im2col of normalized input
    im2col_kernel<<<(Pn * (Kc / 8) + 255) / 256, 256, 0, stream>>>(hcl, A);

    // 4. qkv conv as WMMA GEMM  (M=4608, N=1536, K=4608)
    gemm_bf16_kernel<<<dim3(Pn / 64, N1 / 256), 256, 0, stream>>>(
        A, Wq, qkv_b, nullptr, qkv, N1, 0);

    // 5. cosine scale q,k
    cosine_kernel<<<(Pn * NH * 2) / 8, 256, 0, stream>>>(qkv, scale);

    // 6. neighborhood attention -> bf16 channels-last
    natten_kernel<<<(Pn * NH) / 8, 256, 0, stream>>>(qkv, ocl);

    // 7. im2col of attention output (reuse A)
    im2col_kernel<<<(Pn * (Kc / 8) + 255) / 256, 256, 0, stream>>>(ocl, A);

    // 8. out conv as WMMA GEMM + bias + residual, NCHW epilogue
    gemm_bf16_kernel<<<dim3(Pn / 64, N2 / 256), 256, 0, stream>>>(
        A, Wo, out_b, x, y, N2, 1);
}
